// DiagonalSelectiveSSM_81801947120405
// MI455X (gfx1250) — matrix operand order repaired and checked by reading, not compiled, not run on hardware
//
#include <hip/hip_runtime.h>
#include <hip/hip_bf16.h>
#include <math.h>

// ---------------------------------------------------------------------------
// Problem constants (from reference): B=4, T=8192, D=1024
// ---------------------------------------------------------------------------
#define BB    4
#define TT    8192
#define DD    1024
#define MM    (BB * TT)          // 32768 tokens
#define KK    DD                 // GEMM K
#define NN    DD                 // GEMM N

// GEMM tiling: block = 128(M) x 128(N), 8 waves (4 down M x 2 across N),
// wave tile = 32(M) x 64(N) = 2x4 WMMA 16x16 tiles.
#define TILE_M 128
#define TILE_N 128
#define KC     32                // K per LDS stage (one wmma K)
#define LDAP   40                // padded LDS row stride (halfs) -> conflict-free
#define BUFH   (128 * LDAP)      // halfs per LDS buffer (double-buffered)

// Scan chunking
#define NCHUNK 16
#define CLEN   (TT / NCHUNK)     // 512

typedef _Float16 v16h  __attribute__((ext_vector_type(16)));
typedef float    v8f   __attribute__((ext_vector_type(8)));
typedef float    f32x4 __attribute__((ext_vector_type(4)));
typedef unsigned int u32x4 __attribute__((ext_vector_type(4)));
typedef unsigned int u32x2 __attribute__((ext_vector_type(2)));

union Frag16 { u32x4 q[2]; v16h h; };   // 8 VGPRs = 16 halfs

__device__ __forceinline__ float a_safe_of(float alogit) {
    float a = tanhf(alogit);
    const float eps = 1e-4f;
    if (fabsf(a) < eps) a = (a < 0.0f) ? -eps : eps;
    return a;
}

// Async 16B/lane copy global -> LDS (ASYNCcnt-tracked; no VGPR round trip).
// lds_off = LDS byte address (low 32 bits of the generic pointer to __shared__).
// NOTE: no "memory" clobber here -- ordering is established by async_wait_all()
// (which has one) followed by __syncthreads(); this keeps the unroller happy.
__device__ __forceinline__ void async_copy_b128(const void* gptr, void* lptr) {
    unsigned lds_off = (unsigned)(unsigned long long)lptr;
    asm volatile("global_load_async_to_lds_b128 %0, %1, off"
                 :
                 : "v"(lds_off), "v"(gptr));
}
__device__ __forceinline__ void async_wait_all() {
    asm volatile("s_wait_asynccnt 0x0" ::: "memory");
}

// ---------------------------------------------------------------------------
// Kernel 1: f32 -> f16 conversion (vectorized 4-wide)
// ---------------------------------------------------------------------------
__global__ void ssm_cvt_f16(const float* __restrict__ in,
                            u32x2* __restrict__ out, int n4) {
    int i = blockIdx.x * blockDim.x + threadIdx.x;
    if (i >= n4) return;
    f32x4 v = ((const f32x4*)in)[i];
    union { _Float16 h[4]; u32x2 u; } p;
    p.h[0] = (_Float16)v.x; p.h[1] = (_Float16)v.y;
    p.h[2] = (_Float16)v.z; p.h[3] = (_Float16)v.w;
    out[i] = p.u;
}

// ---------------------------------------------------------------------------
// Kernel 2: fused gate GEMM  u = b * sigmoid(x @ W^T + gate_b) * x
//   xh: [MM][KK] f16 tokens, wh: [NN][KK] f16 gate_W rows (row e, contiguous d)
//   Double-buffered LDS, staged with GLOBAL_LOAD_ASYNC_TO_LDS_B128.
//   K loop fully unrolled -> compile-time buffer parity, no register rotation.
// ---------------------------------------------------------------------------
__global__ __launch_bounds__(256)
void ssm_gate_gemm(const unsigned short* __restrict__ xh,
                   const unsigned short* __restrict__ wh,
                   const float* __restrict__ x,
                   const float* __restrict__ gate_b,
                   const float* __restrict__ bvec,
                   float* __restrict__ u)
{
    __shared__ unsigned short ldsA[2 * BUFH];
    __shared__ unsigned short ldsB[2 * BUFH];

    const int tid   = threadIdx.x;
    const int wave  = tid >> 5;
    const int lane  = tid & 31;
    const int m0    = blockIdx.x * TILE_M;
    const int n0    = blockIdx.y * TILE_N;
    const int waveM = wave & 3;      // 4 waves down M (32 rows each)
    const int waveN = wave >> 2;     // 2 waves across N (64 cols each)

    const int hi = lane >> 4;        // half-wave select
    const int lr = lane & 15;

    v8f acc[2][4] = {};

    // Staging: 128 rows x 32 halfs for A and B (2 async 16B copies each / thread)
    const int srow = tid >> 2;           // 0..63
    const int sc4  = (tid & 3) * 8;      // 0,8,16,24
    const unsigned short* gA = &xh[(size_t)(m0 + srow) * KK + sc4];
    const unsigned short* gB = &wh[(size_t)(n0 + srow) * KK + sc4];
    auto stage = [&](int bufOff, int k0) {
        #pragma unroll
        for (int i = 0; i < 2; ++i) {
            int row = srow + 64 * i;
            async_copy_b128(gA + (size_t)(64 * i) * KK + k0,
                            &ldsA[bufOff + row * LDAP + sc4]);
            async_copy_b128(gB + (size_t)(64 * i) * KK + k0,
                            &ldsB[bufOff + row * LDAP + sc4]);
        }
    };

    const int NK = KK / KC;              // 32
    stage(0, 0);
    async_wait_all();
    __syncthreads();

    #pragma unroll
    for (int kk = 0; kk < NK; ++kk) {
        const int cur = (kk & 1) * BUFH;
        const int nxt = ((kk & 1) ^ 1) * BUFH;
        if (kk + 1 < NK) stage(nxt, (kk + 1) * KC);

        // A fragment: lane m = lr; halfs 0..7 -> K = hi*8.., 8..15 -> K = 16+hi*8..
        Frag16 Af[2], Bf[4];
        #pragma unroll
        for (int mt = 0; mt < 2; ++mt) {
            int row = waveM * 32 + mt * 16 + lr;
            Af[mt].q[0] = *(const u32x4*)&ldsA[cur + row * LDAP + hi * 8];
            Af[mt].q[1] = *(const u32x4*)&ldsA[cur + row * LDAP + 16 + hi * 8];
        }
        // B fragment: lane n = lr; halfs 0..7 -> K = hi*8.., 8..15 -> K = 16+hi*8..
        #pragma unroll
        for (int nt = 0; nt < 4; ++nt) {
            int row = waveN * 64 + nt * 16 + lr;
            Bf[nt].q[0] = *(const u32x4*)&ldsB[cur + row * LDAP + hi * 8];
            Bf[nt].q[1] = *(const u32x4*)&ldsB[cur + row * LDAP + 16 + hi * 8];
        }
        #pragma unroll
        for (int mt = 0; mt < 2; ++mt)
            #pragma unroll
            for (int nt = 0; nt < 4; ++nt)
                acc[mt][nt] = __builtin_amdgcn_wmma_f32_16x16x32_f16(
                    false, Af[mt].h, false, Bf[nt].h,
                    (short)0, acc[mt][nt], false, false);

        // This wave's async copies into `nxt` must land before publishing.
        async_wait_all();
        __syncthreads();
    }

    // Epilogue: sigmoid gate + u = b * g * x
    #pragma unroll
    for (int mt = 0; mt < 2; ++mt) {
        #pragma unroll
        for (int nt = 0; nt < 4; ++nt) {
            int colg = n0 + waveN * 64 + nt * 16 + lr;
            float gb = gate_b[colg];
            float bb = bvec[colg];
            #pragma unroll
            for (int r = 0; r < 8; ++r) {
                int rowg = m0 + waveM * 32 + mt * 16 + r + 8 * hi;
                float logit = acc[mt][nt][r] + gb;
                float g  = 1.0f / (1.0f + __expf(-logit));
                size_t o = (size_t)rowg * NN + colg;
                u[o] = bb * g * x[o];
            }
        }
    }
}

// ---------------------------------------------------------------------------
// Kernel 3a: per-(channel, chunk) partial scan (zero initial state)
//   S_j = sum_{t in chunk} a^{end-t} u_t
// ---------------------------------------------------------------------------
__global__ void ssm_scan_partial(const float* __restrict__ u,
                                 const float* __restrict__ a_logit,
                                 float* __restrict__ Sbuf)
{
    int idx = blockIdx.x * blockDim.x + threadIdx.x;     // 65536 threads
    int dd = idx & (DD - 1);
    int bb = (idx >> 10) & (BB - 1);
    int j  = idx >> 12;
    float a = a_safe_of(a_logit[dd]);
    const float* up = u + ((size_t)(bb * TT + j * CLEN) * DD + dd);
    float S = 0.0f;
    for (int t = 0; t < CLEN; ++t) {
        __builtin_prefetch(up + (size_t)(t + 24) * DD, 0, 1);  // global_prefetch_b8
        S = fmaf(a, S, up[(size_t)t * DD]);
    }
    Sbuf[(size_t)((bb << 10) + dd) * NCHUNK + j] = S;
}

// ---------------------------------------------------------------------------
// Kernel 3b: per-channel carry combine across the 16 chunks (in place)
//   c_0 = 0 ; c_{j+1} = a^CLEN * c_j + S_j  ;  Sbuf[j] <- c_j
// ---------------------------------------------------------------------------
__global__ void ssm_scan_carry(const float* __restrict__ a_logit,
                               float* __restrict__ Sbuf)
{
    int ch = blockIdx.x * blockDim.x + threadIdx.x;      // 4096 threads
    int dd = ch & (DD - 1);
    float a = a_safe_of(a_logit[dd]);
    float aL = a;                                        // a^(2^9) = a^512
    #pragma unroll
    for (int i = 0; i < 9; ++i) aL = aL * aL;
    float c = 0.0f;
    size_t base = (size_t)ch * NCHUNK;
    #pragma unroll
    for (int j = 0; j < NCHUNK; ++j) {
        float S = Sbuf[base + j];
        Sbuf[base + j] = c;
        c = fmaf(aL, c, S);
    }
}

// ---------------------------------------------------------------------------
// Kernel 3c: seeded re-scan, emit h = c*s + d*x
// ---------------------------------------------------------------------------
__global__ void ssm_scan_apply(const float* __restrict__ u,
                               const float* __restrict__ x,
                               const float* __restrict__ a_logit,
                               const float* __restrict__ cvec,
                               const float* __restrict__ dvec,
                               const float* __restrict__ Sbuf,
                               float* __restrict__ out)
{
    int idx = blockIdx.x * blockDim.x + threadIdx.x;     // 65536 threads
    int dd = idx & (DD - 1);
    int bb = (idx >> 10) & (BB - 1);
    int j  = idx >> 12;
    float a  = a_safe_of(a_logit[dd]);
    float cc = cvec[dd];
    float dv = dvec[dd];
    size_t off = (size_t)(bb * TT + j * CLEN) * DD + dd;
    float s = Sbuf[(size_t)((bb << 10) + dd) * NCHUNK + j];  // carry (state at chunk start - 1)
    for (int t = 0; t < CLEN; ++t) {
        size_t o = off + (size_t)t * DD;
        __builtin_prefetch(u + o + (size_t)24 * DD, 0, 1);   // global_prefetch_b8
        __builtin_prefetch(x + o + (size_t)24 * DD, 0, 1);
        float uv = u[o];
        s = fmaf(a, s, uv);
        out[o] = fmaf(cc, s, dv * x[o]);
    }
}

// ---------------------------------------------------------------------------
// Launch
// Inputs (setup_inputs order): 0:x 1:a_logit 2:b 3:c 4:d 5:gate_W 6:gate_b
// Workspace layout:
//   [0,        64MB)  xh   (f16 x)
//   [64MB,     66MB)  wh   (f16 gate_W)
//   [66MB,   +256KB)  Sbuf (carry states)
//   [next,   +128MB)  u    (f32 gated input)
// ---------------------------------------------------------------------------
extern "C" void kernel_launch(void* const* d_in, const int* in_sizes, int n_in,
                              void* d_out, int out_size, void* d_ws, size_t ws_size,
                              hipStream_t stream) {
    (void)in_sizes; (void)n_in; (void)out_size; (void)ws_size;

    const float* x       = (const float*)d_in[0];
    const float* a_logit = (const float*)d_in[1];
    const float* bvec    = (const float*)d_in[2];
    const float* cvec    = (const float*)d_in[3];
    const float* dvec    = (const float*)d_in[4];
    const float* gate_W  = (const float*)d_in[5];
    const float* gate_b  = (const float*)d_in[6];
    float* out = (float*)d_out;

    char* ws = (char*)d_ws;
    const size_t xh_bytes = (size_t)MM * KK * 2;           // 64 MB
    const size_t wh_bytes = (size_t)NN * KK * 2;           // 2 MB
    const size_t sb_bytes = (size_t)BB * DD * NCHUNK * 4;  // 256 KB
    unsigned short* xh = (unsigned short*)ws;
    unsigned short* wh = (unsigned short*)(ws + xh_bytes);
    float*          Sb = (float*)(ws + xh_bytes + wh_bytes);
    float*          u  = (float*)(ws + xh_bytes + wh_bytes + sb_bytes);

    // 1) convert x and gate_W to f16
    {
        int n4x = (MM * KK) / 4;                     // 8388608
        ssm_cvt_f16<<<n4x / 256, 256, 0, stream>>>(x, (u32x2*)xh, n4x);
        int n4w = (NN * KK) / 4;                     // 262144
        ssm_cvt_f16<<<n4w / 256, 256, 0, stream>>>(gate_W, (u32x2*)wh, n4w);
    }

    // 2) fused gate GEMM -> u
    {
        dim3 grid(MM / TILE_M, NN / TILE_N);         // 256 x 8
        ssm_gate_gemm<<<grid, 256, 0, stream>>>(xh, wh, x, gate_b, bvec, u);
    }

    // 3) chunked scan
    {
        int nthr = BB * DD * NCHUNK;                 // 65536
        ssm_scan_partial<<<nthr / 256, 256, 0, stream>>>(u, a_logit, Sb);
        int nch = BB * DD;                           // 4096
        ssm_scan_carry<<<nch / 256, 256, 0, stream>>>(a_logit, Sb);
        ssm_scan_apply<<<nthr / 256, 256, 0, stream>>>(u, x, a_logit, cvec, dvec, Sb, out);
    }
}
